// SelfAttention_31894427140309
// MI455X (gfx1250) — matrix operand order repaired and checked by reading, not compiled, not run on hardware
//
#include <hip/hip_runtime.h>
#include <cstdint>
#include <cstddef>

// ---------------------------------------------------------------------------
// Problem constants (from the reference): B=8, C=256, H=W=64 -> N=4096, CFG=32
// ---------------------------------------------------------------------------
#define BATCH 8
#define CCH   256
#define NPIX  4096
#define CFGD  32

typedef __bf16 bf16_t;
typedef __bf16 v8bf  __attribute__((ext_vector_type(8)));
typedef __bf16 v16bf __attribute__((ext_vector_type(16)));
typedef float  v8f   __attribute__((ext_vector_type(8)));

#define LOG2E 1.44269504088896340736f

__device__ __forceinline__ float fast_exp2(float x) {
#if __has_builtin(__builtin_amdgcn_exp2f)
  return __builtin_amdgcn_exp2f(x);
#else
  return exp2f(x);
#endif
}

__device__ __forceinline__ v8f wmma_bf16(v16bf a, v16bf b, v8f c) {
  // D = A(16x32 bf16) * B(32x16 bf16) + C(16x16 f32)
  return __builtin_amdgcn_wmma_f32_16x16x32_bf16(false, a, false, b,
                                                 (short)0, c, false, false);
}

// Build a 16-element bf16 A/B operand from two contiguous 8-element runs.
__device__ __forceinline__ v16bf join8(v8bf lo, v8bf hi) {
  union { v16bf v; v8bf h[2]; } u;
  u.h[0] = lo; u.h[1] = hi;
  return u.v;
}

// ---------------------------------------------------------------------------
// Kernel 0: convert weights f32 -> bf16.  wfg = [wf(32x256); wg(32x256)],
// whb = wh(256x256).
// ---------------------------------------------------------------------------
__global__ __launch_bounds__(256) void wconv_kernel(
    const float* __restrict__ wf, const float* __restrict__ wg,
    const float* __restrict__ wh, bf16_t* __restrict__ wfg,
    bf16_t* __restrict__ whb) {
  int idx = blockIdx.x * 256 + threadIdx.x;
  if (idx < 64 * 256) {
    int row = idx >> 8, col = idx & 255;
    float v = (row < 32) ? wf[row * 256 + col] : wg[(row - 32) * 256 + col];
    wfg[idx] = (bf16_t)v;
  }
  if (idx < 256 * 256) whb[idx] = (bf16_t)wh[idx];
}

// ---------------------------------------------------------------------------
// Kernel 1: x[b][c][n] f32  ->  xt[b][n][c] bf16 (LDS-tiled 32x32 transpose)
// grid = B * (C/32) * (N/32) = 8192 blocks of 256 threads
// ---------------------------------------------------------------------------
__global__ __launch_bounds__(256) void transpose_kernel(
    const float* __restrict__ x, bf16_t* __restrict__ xt) {
  __shared__ float tile[32][33];
  int b  = blockIdx.x >> 10;
  int ct = (blockIdx.x >> 7) & 7;
  int nt = blockIdx.x & 127;
  int c0 = ct * 32, n0 = nt * 32;
  int tx = threadIdx.x & 31, ty = threadIdx.x >> 5;
#pragma unroll
  for (int p = 0; p < 4; ++p) {
    int c = c0 + ty + 8 * p;
    tile[ty + 8 * p][tx] = x[((size_t)b * CCH + c) * NPIX + n0 + tx];
  }
  __syncthreads();
  int px = threadIdx.x & 15, py = threadIdx.x >> 4;  // px: channel pair
#pragma unroll
  for (int p = 0; p < 2; ++p) {
    int i = py + 16 * p;  // pixel within tile
    union { bf16_t e[2]; unsigned u; } pk;
    pk.e[0] = (bf16_t)tile[2 * px + 0][i];
    pk.e[1] = (bf16_t)tile[2 * px + 1][i];
    *(unsigned*)(xt + ((size_t)b * NPIX + n0 + i) * CCH + c0 + 2 * px) = pk.u;
  }
}

// ---------------------------------------------------------------------------
// Kernel 2: f/g projections.  kbuf[b][j][32] = wf @ x + bf (keys),
// qbuf[b][i][32] = wg @ x + bg (queries), both bf16, pixel-major.
// grid = B * (N/128) = 256 blocks; wave w handles 16 pixels, 4 output tiles.
// ---------------------------------------------------------------------------
__global__ __launch_bounds__(256) void proj_fg_kernel(
    const bf16_t* __restrict__ xt, const bf16_t* __restrict__ wfg,
    const float* __restrict__ bfv, const float* __restrict__ bgv,
    bf16_t* __restrict__ kbuf, bf16_t* __restrict__ qbuf) {
  int lane = threadIdx.x & 31, wave = threadIdx.x >> 5;
  int n15 = lane & 15, hi = lane >> 4;
  int b = blockIdx.x >> 5, pt = blockIdx.x & 31;
  int j0 = pt * 128 + wave * 16;

  const bf16_t* xrow = xt + ((size_t)b * NPIX + j0 + n15) * CCH;
  v8f acc[4];
#pragma unroll
  for (int t = 0; t < 4; ++t) acc[t] = (v8f){0.f,0.f,0.f,0.f,0.f,0.f,0.f,0.f};

  for (int ks = 0; ks < CCH; ks += 32) {
    // B operand: 16 contiguous input channels at this lane's pixel
    v16bf xb = *(const v16bf*)(xrow + ks + 16 * hi);
#pragma unroll
    for (int mt = 0; mt < 4; ++mt) {
      const bf16_t* wrow = wfg + (size_t)(mt * 16 + n15) * CCH + ks;
      v16bf wa = *(const v16bf*)(wrow + 16 * hi);
      acc[mt] = wmma_bf16(wa, xb, acc[mt]);
    }
  }
#pragma unroll
  for (int mt = 0; mt < 4; ++mt) {
    int c0 = (mt & 1) * 16 + 8 * hi;  // first of 8 consecutive out channels
    const float* bias = (mt < 2) ? bfv : bgv;
    bf16_t* dst = (mt < 2) ? kbuf : qbuf;
    union { v8bf v; bf16_t e[8]; } s;
#pragma unroll
    for (int r = 0; r < 8; ++r) s.e[r] = (bf16_t)(acc[mt][r] + bias[c0 + r]);
    *(v8bf*)(dst + ((size_t)b * NPIX + j0 + n15) * CFGD + c0) = s.v;
  }
}

// ---------------------------------------------------------------------------
// Kernel 3: h projection.  vbuf[b][c][j] = wh @ x + bh, bf16, channel-major.
// grid = B * (N/128) * (C/32) = 2048 blocks; wave handles 16 pixels x 32 ch.
// ---------------------------------------------------------------------------
__global__ __launch_bounds__(256) void proj_h_kernel(
    const bf16_t* __restrict__ xt, const bf16_t* __restrict__ whb,
    const float* __restrict__ bhv, bf16_t* __restrict__ vbuf) {
  int lane = threadIdx.x & 31, wave = threadIdx.x >> 5;
  int n15 = lane & 15, hi = lane >> 4;
  int b    = blockIdx.x >> 8;
  int pt   = (blockIdx.x >> 3) & 31;
  int cgrp = blockIdx.x & 7;
  int j0 = pt * 128 + wave * 16;
  int m0 = cgrp * 32;

  const bf16_t* xrow = xt + ((size_t)b * NPIX + j0 + n15) * CCH;
  v8f acc[2];
#pragma unroll
  for (int t = 0; t < 2; ++t) acc[t] = (v8f){0.f,0.f,0.f,0.f,0.f,0.f,0.f,0.f};

  for (int ks = 0; ks < CCH; ks += 32) {
    v16bf xb = *(const v16bf*)(xrow + ks + 16 * hi);
#pragma unroll
    for (int mt = 0; mt < 2; ++mt) {
      const bf16_t* wrow = whb + (size_t)(m0 + mt * 16 + n15) * CCH + ks;
      v16bf wa = *(const v16bf*)(wrow + 16 * hi);
      acc[mt] = wmma_bf16(wa, xb, acc[mt]);
    }
  }
#pragma unroll
  for (int mt = 0; mt < 2; ++mt) {
#pragma unroll
    for (int r = 0; r < 8; ++r) {
      int c = m0 + mt * 16 + r + 8 * hi;
      vbuf[((size_t)b * CCH + c) * NPIX + j0 + n15] =
          (bf16_t)(acc[mt][r] + bhv[c]);
    }
  }
}

// ---------------------------------------------------------------------------
// Kernel 4: flash attention + residual.
// Per wave: 16 queries, O[16x256] in 16 v8f accumulators, streaming softmax
// over 64-key tiles.  K/V (19 MB bf16 total) live in L2 (192 MB).
// grid = B * (N/128) = 256 blocks of 256 threads (8 waves).
// ---------------------------------------------------------------------------
#define LDSROW 20  // padded floats per key-row: kills bank conflicts on reads

__global__ __launch_bounds__(256) void attn_kernel(
    const bf16_t* __restrict__ qbuf, const bf16_t* __restrict__ kbuf,
    const bf16_t* __restrict__ vbuf, const float* __restrict__ x,
    const float* __restrict__ gamma, float* __restrict__ out) {
  __shared__ __align__(16) float plds[8 * 64 * LDSROW];  // 40 KB
  int lane = threadIdx.x & 31, wave = threadIdx.x >> 5;
  int n15 = lane & 15, hi = lane >> 4;
  int b  = blockIdx.x >> 5;
  int qt = blockIdx.x & 31;
  int i0 = qt * 128 + wave * 16;  // this wave's query base
  float* pw = plds + wave * 64 * LDSROW;

  // Resident Q A-operand: row M = query i0+n15, K = 32 channels
  const bf16_t* qrow = qbuf + ((size_t)b * NPIX + i0 + n15) * CFGD;
  v16bf qa = *(const v16bf*)(qrow + 16 * hi);

  v8f acc[16];
#pragma unroll
  for (int t = 0; t < 16; ++t) acc[t] = (v8f){0.f,0.f,0.f,0.f,0.f,0.f,0.f,0.f};
  float m_run[8], l_run[8];
#pragma unroll
  for (int r = 0; r < 8; ++r) { m_run[r] = -1e30f; l_run[r] = 0.f; }

  const bf16_t* kb_base = kbuf + (size_t)b * NPIX * CFGD;
  const bf16_t* vb_base = vbuf + (size_t)b * CCH * NPIX;
  const v8f zero = (v8f){0.f,0.f,0.f,0.f,0.f,0.f,0.f,0.f};

  for (int j0 = 0; j0 < NPIX; j0 += 64) {
    // ---- scores: S[16q x 64k] = Q * K^T (4 WMMAs) ----
    v8f s[4];
#pragma unroll
    for (int t = 0; t < 4; ++t) {
      v16bf kb = *(const v16bf*)(kb_base +
                  (size_t)(j0 + t * 16 + n15) * CFGD + 16 * hi);
      s[t] = wmma_bf16(qa, kb, zero);
    }
    // ---- online softmax (rows r+8*hi live in 16-lane halves) ----
    float scale[8];
#pragma unroll
    for (int r = 0; r < 8; ++r) {
      float a0 = s[0][r] * LOG2E, a1 = s[1][r] * LOG2E;
      float a2 = s[2][r] * LOG2E, a3 = s[3][r] * LOG2E;
      float mx = fmaxf(fmaxf(a0, a1), fmaxf(a2, a3));
      mx = fmaxf(mx, __shfl_xor(mx, 1));
      mx = fmaxf(mx, __shfl_xor(mx, 2));
      mx = fmaxf(mx, __shfl_xor(mx, 4));
      mx = fmaxf(mx, __shfl_xor(mx, 8));
      float mnew = fmaxf(m_run[r], mx);
      scale[r] = fast_exp2(m_run[r] - mnew);
      float p0 = fast_exp2(a0 - mnew), p1 = fast_exp2(a1 - mnew);
      float p2 = fast_exp2(a2 - mnew), p3 = fast_exp2(a3 - mnew);
      s[0][r] = p0; s[1][r] = p1; s[2][r] = p2; s[3][r] = p3;
      float rs = (p0 + p1) + (p2 + p3);
      rs += __shfl_xor(rs, 1);
      rs += __shfl_xor(rs, 2);
      rs += __shfl_xor(rs, 4);
      rs += __shfl_xor(rs, 8);
      l_run[r] = l_run[r] * scale[r] + rs;
      m_run[r] = mnew;
    }
    // ---- rescale accumulators ----
#pragma unroll
    for (int t = 0; t < 16; ++t)
#pragma unroll
      for (int r = 0; r < 8; ++r) acc[t][r] *= scale[r];

    // ---- stage P to LDS column-major [key][row] (C-layout -> A-layout) ----
#pragma unroll
    for (int t = 0; t < 4; ++t) {
      float* dst = pw + (t * 16 + n15) * LDSROW + 8 * hi;
      *(float4*)(dst + 0) = make_float4(s[t][0], s[t][1], s[t][2], s[t][3]);
      *(float4*)(dst + 4) = make_float4(s[t][4], s[t][5], s[t][6], s[t][7]);
    }
    __syncthreads();  // fence (uniform trip count; regions are wave-private)

    v16bf pa0, pa1;
    {
      union { v16bf v; bf16_t e[16]; } u0, u1;
#pragma unroll
      for (int e = 0; e < 16; ++e) {
        int ke = e + 16 * hi;
        u0.e[e] = (bf16_t)pw[ke * LDSROW + n15];
        u1.e[e] = (bf16_t)pw[(ke + 32) * LDSROW + n15];
      }
      pa0 = u0.v; pa1 = u1.v;
    }

    // ---- O += P * V  (32 WMMAs: 16 channel groups x 2 key halves) ----
#pragma unroll
    for (int cg = 0; cg < 16; ++cg) {
      const bf16_t* vrow = vb_base + (size_t)(cg * 16 + n15) * NPIX + j0;
      v16bf vb0 = *(const v16bf*)(vrow + 16 * hi);
      v16bf vb1 = *(const v16bf*)(vrow + 32 + 16 * hi);
      acc[cg] = wmma_bf16(pa0, vb0, acc[cg]);
      acc[cg] = wmma_bf16(pa1, vb1, acc[cg]);
    }
  }

  // ---- epilogue: normalize, residual, store out[b][c][i] ----
  float g = gamma[0];
  float invl[8];
#pragma unroll
  for (int r = 0; r < 8; ++r) invl[r] = 1.0f / l_run[r];
#pragma unroll
  for (int cg = 0; cg < 16; ++cg) {
    int c = cg * 16 + n15;
    const float* xp = x + ((size_t)b * CCH + c) * NPIX;
    float* op = out + ((size_t)b * CCH + c) * NPIX;
#pragma unroll
    for (int r = 0; r < 8; ++r) {
      int i = i0 + r + 8 * hi;
      op[i] = g * (acc[cg][r] * invl[r]) + xp[i];
    }
  }
}

// ---------------------------------------------------------------------------
// Host launcher.  Workspace layout (bytes):
//   xt   @ 0         : B*N*C*2   = 16,777,216
//   kbuf @ 16777216  : B*N*32*2  =  2,097,152
//   qbuf @ 18874368  : B*N*32*2  =  2,097,152
//   vbuf @ 20971520  : B*C*N*2   = 16,777,216
//   wfg  @ 37748736  : 64*256*2  =     32,768
//   whb  @ 37781504  : 256*256*2 =    131,072   (total ~38 MB)
// ---------------------------------------------------------------------------
extern "C" void kernel_launch(void* const* d_in, const int* in_sizes, int n_in,
                              void* d_out, int out_size, void* d_ws,
                              size_t ws_size, hipStream_t stream) {
  const float* x     = (const float*)d_in[0];
  const float* wf    = (const float*)d_in[1];
  const float* bfv   = (const float*)d_in[2];
  const float* wg    = (const float*)d_in[3];
  const float* bgv   = (const float*)d_in[4];
  const float* wh    = (const float*)d_in[5];
  const float* bhv   = (const float*)d_in[6];
  const float* gamma = (const float*)d_in[7];
  float* out = (float*)d_out;

  char* ws = (char*)d_ws;
  bf16_t* xt   = (bf16_t*)(ws + 0);
  bf16_t* kbuf = (bf16_t*)(ws + 16777216);
  bf16_t* qbuf = (bf16_t*)(ws + 18874368);
  bf16_t* vbuf = (bf16_t*)(ws + 20971520);
  bf16_t* wfg  = (bf16_t*)(ws + 37748736);
  bf16_t* whb  = (bf16_t*)(ws + 37781504);

  wconv_kernel<<<256, 256, 0, stream>>>(wf, wg, wh, wfg, whb);
  transpose_kernel<<<BATCH * (CCH / 32) * (NPIX / 32), 256, 0, stream>>>(x, xt);
  proj_fg_kernel<<<BATCH * (NPIX / 128), 256, 0, stream>>>(xt, wfg, bfv, bgv,
                                                           kbuf, qbuf);
  proj_h_kernel<<<BATCH * (NPIX / 128) * (CCH / 32), 256, 0, stream>>>(
      xt, whb, bhv, vbuf);
  attn_kernel<<<BATCH * (NPIX / 128), 256, 0, stream>>>(qbuf, kbuf, vbuf, x,
                                                        gamma, out);
}
